// MultiHeadAttention_69277822484632
// MI455X (gfx1250) — matrix operand order repaired and checked by reading, not compiled, not run on hardware
//
#include <hip/hip_runtime.h>
#include <hip/hip_bf16.h>
#include <math.h>

// ---------------------------------------------------------------------------
// MI455X (gfx1250) fused multi-head attention:
//   1) QKV projection GEMM (f32 -> f16, f32 WMMA accumulate).
//      Q pre-scaled by hs^-0.5; V written transposed per head [B,H,64,2048].
//   2) flash attention: K/V tiles staged by the Tensor Data Mover
//      (tensor_load_to_lds, TENSORcnt double-buffered), online softmax,
//      8 waves / 128 query rows per block, Q fragments hoisted.
//   3) output projection GEMM (f16 -> f32 + bias).
// Matmuls: v_wmma_f32_16x16x32_f16 (wave32).
// ---------------------------------------------------------------------------

typedef __attribute__((ext_vector_type(16))) _Float16 v16h;
typedef __attribute__((ext_vector_type(8)))  _Float16 v8h;
typedef __attribute__((ext_vector_type(8)))  float    v8f;
typedef unsigned int __attribute__((ext_vector_type(4))) v4u;
typedef int          __attribute__((ext_vector_type(8))) v8i;
typedef int          __attribute__((ext_vector_type(4))) v4i;

static __device__ __forceinline__ v16h make_frag(const _Float16* p0,
                                                 const _Float16* p1) {
  union { v16h v; v8h h[2]; } u;
  u.h[0] = *(const v8h*)p0;   // ds_load_b128
  u.h[1] = *(const v8h*)p1;   // ds_load_b128
  return u.v;
}

static __device__ __forceinline__ v8f wmma_f16(v16h a, v16h b, v8f c) {
  return __builtin_amdgcn_wmma_f32_16x16x32_f16(false, a, false, b,
                                                (short)0, c, false, false);
}

static __device__ __forceinline__ unsigned pack2(float x, float y) {
  union { _Float16 h[2]; unsigned u; } p;
  p.h[0] = (_Float16)x; p.h[1] = (_Float16)y;
  return p.u;
}

// TDM 2D tile load: tile_d0 contiguous f16 elements per row, tile_d1 rows,
// row stride stride_elems, destination LDS rows padded 64->72 halves
// (pad_interval = 32 DWORDs -> code 4, pad_amount = 4 DWORDs -> code 3).
// Descriptor layout per CDNA5 ISA.  This toolchain's
// builtin is the 6-arg clang-23 form: (v4u, v8i, v4i, v4i, v8i, i32 cpol).
static __device__ __forceinline__ void tdm_load_2d(const _Float16* lds_dst,
                                                   const _Float16* gsrc,
                                                   unsigned stride_elems,
                                                   unsigned tile_d0,
                                                   unsigned tile_d1) {
  unsigned long long ga = (unsigned long long)(uintptr_t)gsrc;
  unsigned lds_addr = (unsigned)(uintptr_t)lds_dst;   // low 32 bits = LDS offset
  v4u g0;
  g0[0] = 1u;                                          // count = 1 (valid D#)
  g0[1] = lds_addr;                                    // lds_addr [63:32]
  g0[2] = (unsigned)(ga & 0xffffffffull);              // global_addr [95:64]
  g0[3] = (unsigned)((ga >> 32) & 0x01ffffffull)       // global_addr [120:96]
        | (2u << 30);                                  // type = 2 ("image")
  const unsigned td0 = 1u << 20, td1 = 1u << 20;       // generous tensor dims
  v8i g1;
  g1[0] = (int)((1u << 16)     // data_size = 2 bytes
              | (1u << 20)     // pad_enable
              | (4u << 22)     // pad_interval: 32 DWORDs
              | (3u << 25));   // pad_amount: 4 DWORDs
  g1[1] = (int)((td0 & 0xffffu) << 16);                            // dim0 lo16
  g1[2] = (int)(((td0 >> 16) & 0xffffu) | ((td1 & 0xffffu) << 16));// dim0 hi/dim1 lo
  g1[3] = (int)(((td1 >> 16) & 0xffffu) | (tile_d0 << 16));        // tile_dim0
  g1[4] = (int)(tile_d1 & 0xffffu);                                // tile_dim1
  g1[5] = (int)stride_elems;                                       // dim0_stride
  g1[6] = 0;
  g1[7] = 0;
  v4i z4 = {0, 0, 0, 0};                               // 2D: groups 2/3 unused
  v8i z8 = {0, 0, 0, 0, 0, 0, 0, 0};
  __builtin_amdgcn_tensor_load_to_lds(g0, g1, z4, z4, z8, 0);
}

// Problem constants
constexpr int Dm   = 768;
constexpr int Bsz  = 4;
constexpr int Seq  = 2048;
constexpr int H    = 12;
constexpr int HS   = 64;
constexpr int Mrow = Bsz * Seq;   // 8192

// ---------------------------------------------------------------------------
// GEMM: Y[z] = A @ W[z] + b[z].  128x64 tile / 128 threads (4 waves); each
// wave owns 32x64 (2 A-frags x 4 B-frags = 8 WMMA per K-step of 32).
// QKV mode: z==0 scales output by 0.125; z==2 stores V as [B,H,64,2048].
// ---------------------------------------------------------------------------
template<bool AHALF, bool OUTF32, bool QKV>
__global__ void __launch_bounds__(128) gemm_bias(
    const void* __restrict__ Ap,
    const float* __restrict__ W0, const float* __restrict__ W1,
    const float* __restrict__ W2,
    const float* __restrict__ b0, const float* __restrict__ b1,
    const float* __restrict__ b2,
    void* __restrict__ Y0, void* __restrict__ Y1, void* __restrict__ Y2) {
  const int z = QKV ? blockIdx.z : 0;
  const float* W    = (z == 0) ? W0 : (z == 1 ? W1 : W2);
  const float* bias = (z == 0) ? b0 : (z == 1 ? b1 : b2);
  void*        Y    = (z == 0) ? Y0 : (z == 1 ? Y1 : Y2);

  constexpr int LDT = 40;                       // 80B rows (16B aligned)
  __shared__ __align__(16) _Float16 As[128 * LDT];
  __shared__ __align__(16) _Float16 Bs[64 * LDT];

  const int t    = threadIdx.x;
  const int lane = t & 31;
  const int wave = t >> 5;
  const int lx   = lane & 15;
  const int grp  = lane >> 4;
  const int n0   = blockIdx.x * 64;
  const int m0   = blockIdx.y * 128;
  const int kbA  = (lane < 16) ? 0 : 8;
  const int kbB  = (lane < 16) ? 0 : 8;

  v8f acc[2][4] = {};

  for (int k0 = 0; k0 < Dm; k0 += 32) {
    // ---- stage A tile (128 x 32) as f16 ----
    if constexpr (AHALF) {
      const _Float16* A = (const _Float16*)Ap;
#pragma unroll
      for (int i = 0; i < 4; ++i) {
        int e = t + 128 * i;                     // 512 x v8h
        int r = e >> 2, c8 = e & 3;
        *(v8h*)&As[r * LDT + c8 * 8] =
            *(const v8h*)&A[(size_t)(m0 + r) * Dm + k0 + c8 * 8];
      }
    } else {
      const float* A = (const float*)Ap;
#pragma unroll
      for (int i = 0; i < 8; ++i) {
        int e = t + 128 * i;                     // 1024 x float4
        int r = e >> 3, c4 = e & 7;
        float4 f = *(const float4*)&A[(size_t)(m0 + r) * Dm + k0 + c4 * 4];
        unsigned* d = (unsigned*)&As[r * LDT + c4 * 4];
        d[0] = pack2(f.x, f.y);                  // ds_store_b32 x2
        d[1] = pack2(f.z, f.w);
      }
    }
    // ---- stage W tile transposed: Bs[n][k], packed half2 stores ----
#pragma unroll
    for (int i = 0; i < 2; ++i) {
      int e  = t + 128 * i;                      // 16 k-pairs x 16 n-quads
      int kp = e >> 4, c4 = e & 15;
      int k  = kp * 2;
      float4 f0 = *(const float4*)&W[(size_t)(k0 + k)     * Dm + n0 + c4 * 4];
      float4 f1 = *(const float4*)&W[(size_t)(k0 + k + 1) * Dm + n0 + c4 * 4];
      *(unsigned*)&Bs[(c4 * 4 + 0) * LDT + k] = pack2(f0.x, f1.x);
      *(unsigned*)&Bs[(c4 * 4 + 1) * LDT + k] = pack2(f0.y, f1.y);
      *(unsigned*)&Bs[(c4 * 4 + 2) * LDT + k] = pack2(f0.z, f1.z);
      *(unsigned*)&Bs[(c4 * 4 + 3) * LDT + k] = pack2(f0.w, f1.w);
    }
    if (k0 + 32 < Dm)                            // global_prefetch_b8
      __builtin_prefetch(&W[(size_t)(k0 + 32) * Dm + n0 + (t & 63) * 4], 0, 0);
    __syncthreads();

    v16h a[2];
#pragma unroll
    for (int i = 0; i < 2; ++i) {
      int row = wave * 32 + i * 16 + lx;
      a[i] = make_frag(&As[row * LDT + kbA], &As[row * LDT + kbA + 16]);
    }
#pragma unroll
    for (int nt = 0; nt < 4; ++nt) {
      int col = nt * 16 + lx;
      v16h b = make_frag(&Bs[col * LDT + kbB], &Bs[col * LDT + kbB + 16]);
      acc[0][nt] = wmma_f16(a[0], b, acc[0][nt]);
      acc[1][nt] = wmma_f16(a[1], b, acc[1][nt]);
    }
    __syncthreads();
  }

  // ---- epilogue ----
#pragma unroll
  for (int i = 0; i < 2; ++i) {
    const int rbase = m0 + wave * 32 + i * 16 + 8 * grp;
#pragma unroll
    for (int nt = 0; nt < 4; ++nt) {
      int col = n0 + nt * 16 + lx;
      float bv = bias[col];
#pragma unroll
      for (int v = 0; v < 8; ++v) {
        float val = acc[i][nt][v] + bv;
        int row = rbase + v;
        if constexpr (OUTF32) {
          ((float*)Y)[(size_t)row * Dm + col] = val;
        } else {
          if (QKV && z == 0) val *= 0.125f;      // fold hs^-0.5 into Q
          if (QKV && z == 2) {                   // V -> [B,H,64,2048]
            int bb = row >> 11, s = row & 2047;  // Seq = 2048
            int hh = col >> 6,  d = col & 63;    // HS  = 64
            ((_Float16*)Y)[(((size_t)(bb * H + hh)) * HS + d) * Seq + s] =
                (_Float16)val;
          } else {
            ((_Float16*)Y)[(size_t)row * Dm + col] = (_Float16)val;
          }
        }
      }
    }
  }
}

// ---------------------------------------------------------------------------
// Flash attention.  8 waves / 256 threads = one (b, h, 128-row query tile).
// Q and double-buffered K/V tiles are staged by the Tensor Data Mover; wave 0
// issues the next tile's loads, waits TENSORcnt <= 2, then a workgroup
// barrier releases compute on the current tile (copy/compute overlap).
// ---------------------------------------------------------------------------
__global__ void __launch_bounds__(256) attn_kernel(
    const _Float16* __restrict__ Qg, const _Float16* __restrict__ Kg,
    const _Float16* __restrict__ Vtg, _Float16* __restrict__ Og) {
  constexpr int LDQ = 72;                        // 64 + 8 pad (TDM pad config)
  __shared__ __align__(16) _Float16 Qs[128 * LDQ];
  __shared__ __align__(16) _Float16 Ks[2][64 * LDQ];     // [key][d]
  __shared__ __align__(16) _Float16 Vt[2][64 * LDQ];     // [d][key]
  __shared__ __align__(16) _Float16 Ps[8 * 16 * LDQ];    // per-wave P tile

  const int t    = threadIdx.x;
  const int lane = t & 31;
  const int wave = t >> 5;
  const int lx   = lane & 15;
  const int grp  = lane >> 4;
  const int kbA  = (lane < 16) ? 0 : 8;
  const int kbB  = (lane < 16) ? 0 : 8;

  const int    h       = blockIdx.y;
  const int    b       = blockIdx.z;
  const size_t rowBase = (size_t)b * Seq;
  const int    q0      = blockIdx.x * 128;
  const int    hofs    = h * HS;
  const size_t vBase   = ((size_t)(b * H + h)) * HS * Seq;   // [B,H,64,2048]
  constexpr int NT     = Seq / 64;               // 32 key tiles

  // ---- TDM: stage Q (128x64) and the first K/V tiles ----
  if (wave == 0) {
    tdm_load_2d(&Qs[0], &Qg[(rowBase + q0) * Dm + hofs], Dm, HS, 128);
    tdm_load_2d(&Ks[0][0], &Kg[rowBase * Dm + hofs], Dm, HS, 64);
    tdm_load_2d(&Vt[0][0], &Vtg[vBase], Seq, 64, HS);
    __builtin_amdgcn_s_wait_tensorcnt(0);
  }
  __syncthreads();

  // Q fragments are loop-invariant: load once.
  const int qrow = wave * 16 + lx;
  v16h qa0 = make_frag(&Qs[qrow * LDQ + kbA],      &Qs[qrow * LDQ + kbA + 16]);
  v16h qa1 = make_frag(&Qs[qrow * LDQ + 32 + kbA], &Qs[qrow * LDQ + 32 + kbA + 16]);

  float m[8], l[8];
#pragma unroll
  for (int v = 0; v < 8; ++v) { m[v] = -INFINITY; l[v] = 0.0f; }
  v8f o[4] = {};
  _Float16* Pw = &Ps[wave * 16 * LDQ];

  for (int kt = 0; kt < NT; ++kt) {
    const int cur = kt & 1;
    // ---- TDM: prefetch next K/V tile into the other buffer ----
    if (wave == 0) {
      if (kt + 1 < NT) {
        const int kbn = (kt + 1) * 64;
        tdm_load_2d(&Ks[cur ^ 1][0], &Kg[(rowBase + kbn) * Dm + hofs], Dm, HS, 64);
        tdm_load_2d(&Vt[cur ^ 1][0], &Vtg[vBase + kbn], Seq, 64, HS);
        __builtin_amdgcn_s_wait_tensorcnt(2);    // current tile complete
      } else {
        __builtin_amdgcn_s_wait_tensorcnt(0);
      }
    }
    __syncthreads();                             // release compute on cur
    const _Float16* Kc = &Ks[cur][0];
    const _Float16* Vc = &Vt[cur][0];

    // ---- S = Q @ K^T (Q pre-scaled by 0.125 in projection) ----
    v8f s[4];
#pragma unroll
    for (int nt = 0; nt < 4; ++nt) {
      int kcol = nt * 16 + lx;
      v16h b0 = make_frag(&Kc[kcol * LDQ + kbB],      &Kc[kcol * LDQ + kbB + 16]);
      v16h b1 = make_frag(&Kc[kcol * LDQ + 32 + kbB], &Kc[kcol * LDQ + 32 + kbB + 16]);
      v8f zf = {};
      zf = wmma_f16(qa0, b0, zf);
      zf = wmma_f16(qa1, b1, zf);
      s[nt] = zf;
    }

    // ---- online softmax (row reduction over 16-lane half-group) ----
    float alpha[8];
#pragma unroll
    for (int v = 0; v < 8; ++v) {
      float x = fmaxf(fmaxf(s[0][v], s[1][v]), fmaxf(s[2][v], s[3][v]));
      x = fmaxf(x, __shfl_xor(x, 1, 32));
      x = fmaxf(x, __shfl_xor(x, 2, 32));
      x = fmaxf(x, __shfl_xor(x, 4, 32));
      x = fmaxf(x, __shfl_xor(x, 8, 32));
      float mn = fmaxf(m[v], x);
      alpha[v] = __expf(m[v] - mn);
      m[v] = mn;
    }
    float rs[8];
#pragma unroll
    for (int v = 0; v < 8; ++v) rs[v] = 0.0f;
#pragma unroll
    for (int nt = 0; nt < 4; ++nt) {
      int col = nt * 16 + lx;
#pragma unroll
      for (int v = 0; v < 8; ++v) {
        float p = __expf(s[nt][v] - m[v]);
        rs[v] += p;
        Pw[(v + 8 * grp) * LDQ + col] = (_Float16)p;
      }
    }
#pragma unroll
    for (int v = 0; v < 8; ++v) {
      float x = rs[v];
      x += __shfl_xor(x, 1, 32);
      x += __shfl_xor(x, 2, 32);
      x += __shfl_xor(x, 4, 32);
      x += __shfl_xor(x, 8, 32);
      l[v] = l[v] * alpha[v] + x;
    }
#pragma unroll
    for (int nt = 0; nt < 4; ++nt)
#pragma unroll
      for (int v = 0; v < 8; ++v) o[nt][v] *= alpha[v];

    // ---- O += P @ V ----
    v16h pa0 = make_frag(&Pw[lx * LDQ + kbA],      &Pw[lx * LDQ + kbA + 16]);
    v16h pa1 = make_frag(&Pw[lx * LDQ + 32 + kbA], &Pw[lx * LDQ + 32 + kbA + 16]);
#pragma unroll
    for (int nt = 0; nt < 4; ++nt) {
      int vcol = nt * 16 + lx;
      v16h vb0 = make_frag(&Vc[vcol * LDQ + kbB],      &Vc[vcol * LDQ + kbB + 16]);
      v16h vb1 = make_frag(&Vc[vcol * LDQ + 32 + kbB], &Vc[vcol * LDQ + 32 + kbB + 16]);
      o[nt] = wmma_f16(pa0, vb0, o[nt]);
      o[nt] = wmma_f16(pa1, vb1, o[nt]);
    }
    __syncthreads();                             // done reading cur buffers
  }

  // ---- finalize: O / l -> f16 workspace [B,S,H*hs] ----
#pragma unroll
  for (int nt = 0; nt < 4; ++nt) {
    int col = hofs + nt * 16 + lx;
#pragma unroll
    for (int v = 0; v < 8; ++v) {
      float val = o[nt][v] / l[v];
      Og[(rowBase + q0 + wave * 16 + 8 * grp + v) * Dm + col] = (_Float16)val;
    }
  }
}

// ---------------------------------------------------------------------------
extern "C" void kernel_launch(void* const* d_in, const int* in_sizes, int n_in,
                              void* d_out, int out_size, void* d_ws,
                              size_t ws_size, hipStream_t stream) {
  const float* x  = (const float*)d_in[0];
  const float* Wq = (const float*)d_in[1];
  const float* bq = (const float*)d_in[2];
  const float* Wk = (const float*)d_in[3];
  const float* bk = (const float*)d_in[4];
  const float* Wv = (const float*)d_in[5];
  const float* bv = (const float*)d_in[6];
  const float* Wo = (const float*)d_in[7];
  const float* bo = (const float*)d_in[8];

  const size_t NE = (size_t)Mrow * Dm;           // elements per f16 buffer
  _Float16* Qg = (_Float16*)d_ws;
  _Float16* Kg = Qg + NE;
  _Float16* Vg = Kg + NE;                        // holds [B,H,64,2048]
  _Float16* Og = Vg + NE;

  // 1) fused QKV projections (z: 0=Q scaled, 1=K, 2=V transposed)
  dim3 gq(Dm / 64, Mrow / 128, 3);
  gemm_bias<false, false, true><<<gq, 128, 0, stream>>>(
      x, Wq, Wk, Wv, bq, bk, bv, (void*)Qg, (void*)Kg, (void*)Vg);

  // 2) flash attention: (128-row qtile, head, batch)
  dim3 ga(Seq / 128, H, Bsz);
  attn_kernel<<<ga, 256, 0, stream>>>(Qg, Kg, Vg, Og);

  // 3) output projection -> f32 d_out
  dim3 go(Dm / 64, Mrow / 128, 1);
  gemm_bias<true, true, false><<<go, 128, 0, stream>>>(
      (const void*)Og, Wo, Wo, Wo, bo, bo, bo, d_out, d_out, d_out);
}
